// BigGNN_49228915146752
// MI455X (gfx1250) — compile-verified, matrix operand order checked by reading, not run on hardware
//
#include <hip/hip_runtime.h>
#include <hip/hip_bf16.h>
#include <math.h>

typedef _Float16 v8h  __attribute__((ext_vector_type(8)));
typedef _Float16 v16h __attribute__((ext_vector_type(16)));
typedef float    v8f  __attribute__((ext_vector_type(8)));

#define D      300
#define DPAD   304   // padded feature dim (rows of B / cols of C)
#define KPAD   320   // padded K for x@W GEMMs (multiple of 32)
#define NEG_SLOPE 0.2f

__device__ __forceinline__ float lrelu(float x) { return x > 0.f ? x : NEG_SLOPE * x; }

__device__ __forceinline__ float wave_sum(float v) {
  #pragma unroll
  for (int o = 16; o > 0; o >>= 1) v += __shfl_xor(v, o, 32);
  return v;
}

__device__ __forceinline__ void atomicMaxF(float* addr, float val) {
  int* ia = (int*)addr;
  int old = __float_as_int(*addr);
  while (__int_as_float(old) < val) {
    int assumed = old;
    old = atomicCAS(ia, assumed, __float_as_int(val));
    if (old == assumed) break;
  }
}

// ---------------------------------------------------------------------------
// WMMA GEMM:  C[M x Nreal] = A[M x Kp] (f16, row-major) * B^T
// B supplied transposed: B[n][k], n < Ntiles*16 rows allocated, k < Kp.
// One wave per 16x16 output tile. Kp must be a multiple of 32, M of 16.
// ---------------------------------------------------------------------------
__global__ __launch_bounds__(128)
void wmma_gemm(const _Float16* __restrict__ A, const _Float16* __restrict__ B,
               float* __restrict__ C, int Nreal, int Ntiles, int Kp, int ldc)
{
  int nt = blockIdx.y * blockDim.y + threadIdx.y;
  if (nt >= Ntiles) return;                     // whole wave exits together
  int mt   = blockIdx.x;
  int lane = threadIdx.x;
  int mrow = mt * 16 + (lane & 15);
  int nrow = nt * 16 + (lane & 15);
  int koff = (lane & 16) ? 8 : 0;               // CDNA5 16-bit A/B operand layout
  const _Float16* aBase = A + (size_t)mrow * Kp + koff;
  const _Float16* bBase = B + (size_t)nrow * Kp + koff;

  v8f acc = {};
  for (int kk = 0; kk < Kp; kk += 32) {
    const v8h* ap = (const v8h*)(aBase + kk);
    const v8h* bp = (const v8h*)(bBase + kk);
    v8h a0 = ap[0], a1 = ap[2];                 // K = kk+koff..+7 and +16..+23
    v8h b0 = bp[0], b1 = bp[2];
    if (kk + 32 < Kp) {                         // gfx1250 global_prefetch path
      __builtin_prefetch(aBase + kk + 64, 0, 1);
      __builtin_prefetch(bBase + kk + 64, 0, 1);
    }
    v16h av = __builtin_shufflevector(a0, a1, 0,1,2,3,4,5,6,7,8,9,10,11,12,13,14,15);
    v16h bv = __builtin_shufflevector(b0, b1, 0,1,2,3,4,5,6,7,8,9,10,11,12,13,14,15);
    acc = __builtin_amdgcn_wmma_f32_16x16x32_f16(false, av, false, bv,
                                                 (short)0, acc, false, false);
  }
  int nc = nt * 16 + (lane & 15);
  if (nc < Nreal) {
    int mbase = mt * 16 + ((lane & 16) ? 8 : 0);
    #pragma unroll
    for (int v = 0; v < 8; ++v)
      C[(size_t)(mbase + v) * ldc + nc] = acc[v];
  }
}

// ---------------------------------------------------------------------------
// Conversions
// ---------------------------------------------------------------------------
__global__ void convert_W(const float* __restrict__ W, _Float16* __restrict__ Wh)
{ // Wh[n][k] = W[k][n], padded to DPAD x KPAD
  int idx = blockIdx.x * 256 + threadIdx.x;
  if (idx >= DPAD * KPAD) return;
  int n = idx / KPAD, k = idx % KPAD;
  Wh[idx] = (n < D && k < D) ? (_Float16)W[(size_t)k * D + n] : (_Float16)0.f;
}

__global__ void convert_x(const float* __restrict__ x, _Float16* __restrict__ xh,
                          int rows, int rowoff)
{ // x packed stride D -> xh stride KPAD, zero padded
  int idx = blockIdx.x * 256 + threadIdx.x;
  if (idx >= rows * KPAD) return;
  int r = idx / KPAD, k = idx % KPAD;
  xh[(size_t)(rowoff + r) * KPAD + k] = (k < D) ? (_Float16)x[(size_t)r * D + k]
                                                : (_Float16)0.f;
}

__global__ void transpose_z(const float* __restrict__ z, _Float16* __restrict__ zT,
                            int Nsrc, int srcbase)
{ // zT[f][i] = z[srcbase+i][f], f padded to DPAD rows
  int idx = blockIdx.x * 256 + threadIdx.x;
  if (idx >= DPAD * Nsrc) return;
  int f = idx / Nsrc, i = idx % Nsrc;
  zT[idx] = (f < D) ? (_Float16)z[(size_t)(srcbase + i) * DPAD + f] : (_Float16)0.f;
}

// ---------------------------------------------------------------------------
// Row ops
// ---------------------------------------------------------------------------
__global__ void l2norm_inplace(float* __restrict__ x)
{ // wave per row, packed stride D
  int row = blockIdx.x, lane = threadIdx.x;
  float ss = 0.f;
  for (int f = lane; f < D; f += 32) { float v = x[(size_t)row * D + f]; ss += v * v; }
  ss = wave_sum(ss);
  float inv = 1.0f / fmaxf(sqrtf(ss), 1e-12f);
  for (int f = lane; f < D; f += 32) x[(size_t)row * D + f] *= inv;
}

__global__ void rowdot(const float* __restrict__ z, const float* __restrict__ as,
                       const float* __restrict__ ad, float* __restrict__ es,
                       float* __restrict__ ed)
{ // wave per row of z (stride DPAD)
  int row = blockIdx.x, lane = threadIdx.x;
  float s1 = 0.f, s2 = 0.f;
  for (int f = lane; f < D; f += 32) {
    float zv = z[(size_t)row * DPAD + f];
    s1 += zv * as[f];  s2 += zv * ad[f];
  }
  s1 = wave_sum(s1); s2 = wave_sum(s2);
  if (lane == 0) { es[row] = s1; ed[row] = s2; }
}

// ---------------------------------------------------------------------------
// Sparse GAT (edge-parallel)
// ---------------------------------------------------------------------------
__global__ void sparse_init(float* __restrict__ agg, float* __restrict__ emax,
                            float* __restrict__ denom, const float* __restrict__ b)
{ // block of 320 per row
  int row = blockIdx.x, tid = threadIdx.x;
  if (tid < DPAD) agg[(size_t)row * DPAD + tid] = (tid < D) ? b[tid] : 0.f;
  if (tid == DPAD) { emax[row] = -1e30f; denom[row] = 0.f; }
}

__global__ void edge_max(const int* __restrict__ src, const int* __restrict__ dst,
                         int E, int N, const float* __restrict__ es,
                         const float* __restrict__ ed, float* __restrict__ ebuf,
                         float* __restrict__ emax)
{
  int idx = blockIdx.x * 256 + threadIdx.x;
  if (idx >= E + N) return;
  int s = (idx < E) ? src[idx] : idx - E;
  int d = (idx < E) ? dst[idx] : idx - E;
  float e = lrelu(es[s] + ed[d]);
  ebuf[idx] = e;
  atomicMaxF(&emax[d], e);
}

__global__ void edge_exp(const int* __restrict__ src, const int* __restrict__ dst,
                         int E, int N, const float* __restrict__ ebuf,
                         const float* __restrict__ emax, float* __restrict__ exbuf,
                         float* __restrict__ denom)
{
  int idx = blockIdx.x * 256 + threadIdx.x;
  if (idx >= E + N) return;
  int d = (idx < E) ? dst[idx] : idx - E;
  float ex = expf(ebuf[idx] - emax[d]);
  exbuf[idx] = ex;
  atomicAdd(&denom[d], ex);
}

__global__ void edge_agg(const int* __restrict__ src, const int* __restrict__ dst,
                         int E, const float* __restrict__ exbuf,
                         const float* __restrict__ denom, const float* __restrict__ z,
                         float* __restrict__ agg)
{ // block (128) per edge
  int idx = blockIdx.x, tid = threadIdx.x;
  int s = (idx < E) ? src[idx] : idx - E;
  int d = (idx < E) ? dst[idx] : idx - E;
  float alpha = exbuf[idx] / denom[d];
  for (int f = tid; f < D; f += 128)
    atomicAdd(&agg[(size_t)d * DPAD + f], alpha * z[(size_t)s * DPAD + f]);
}

__global__ void finish_sparse(const float* __restrict__ agg, float* __restrict__ xout)
{ // wave per row: relu + l2norm, agg stride DPAD -> xout stride D
  int row = blockIdx.x, lane = threadIdx.x;
  float ss = 0.f;
  for (int f = lane; f < D; f += 32) {
    float v = fmaxf(agg[(size_t)row * DPAD + f], 0.f);
    ss += v * v;
  }
  ss = wave_sum(ss);
  float inv = 1.0f / fmaxf(sqrtf(ss), 1e-12f);
  for (int f = lane; f < D; f += 32) {
    float v = fmaxf(agg[(size_t)row * DPAD + f], 0.f);
    xout[(size_t)row * D + f] = v * inv;
  }
}

// ---------------------------------------------------------------------------
// Dense cross-GAT softmax (fully bipartite + self loop)
// ---------------------------------------------------------------------------
__global__ void cross_softmax(const float* __restrict__ es, const float* __restrict__ ed,
                              _Float16* __restrict__ alphaH, float* __restrict__ aself,
                              int Nsrc, int srcbase)
{ // block (256) per dst row j
  int j = blockIdx.x, tid = threadIdx.x;
  __shared__ float red[256];
  float edj = ed[j];
  float eself = lrelu(es[j] + edj);
  float m = eself;
  for (int i = tid; i < Nsrc; i += 256)
    m = fmaxf(m, lrelu(es[srcbase + i] + edj));
  red[tid] = m; __syncthreads();
  for (int s = 128; s > 0; s >>= 1) { if (tid < s) red[tid] = fmaxf(red[tid], red[tid + s]); __syncthreads(); }
  m = red[0]; __syncthreads();

  float sum = 0.f;
  for (int i = tid; i < Nsrc; i += 256) {
    float ex = expf(lrelu(es[srcbase + i] + edj) - m);
    alphaH[(size_t)j * Nsrc + i] = (_Float16)ex;
    sum += ex;
  }
  if (tid == 0) sum += expf(eself - m);
  red[tid] = sum; __syncthreads();
  for (int s = 128; s > 0; s >>= 1) { if (tid < s) red[tid] += red[tid + s]; __syncthreads(); }
  float inv = 1.0f / red[0];

  for (int i = tid; i < Nsrc; i += 256) {
    float ex = (float)alphaH[(size_t)j * Nsrc + i];
    alphaH[(size_t)j * Nsrc + i] = (_Float16)(ex * inv);
  }
  if (tid == 0) aself[j] = expf(eself - m) * inv;
}

__global__ void cross_finish(const float* __restrict__ gout, const float* __restrict__ z,
                             const float* __restrict__ aself, const float* __restrict__ b,
                             float* __restrict__ xout)
{ // wave per dst row: add self term + bias, relu, l2norm
  int row = blockIdx.x, lane = threadIdx.x;
  float a = aself[row];
  float ss = 0.f;
  for (int f = lane; f < D; f += 32) {
    float v = gout[(size_t)row * DPAD + f] + a * z[(size_t)row * DPAD + f] + b[f];
    v = fmaxf(v, 0.f);
    ss += v * v;
  }
  ss = wave_sum(ss);
  float inv = 1.0f / fmaxf(sqrtf(ss), 1e-12f);
  for (int f = lane; f < D; f += 32) {
    float v = gout[(size_t)row * DPAD + f] + a * z[(size_t)row * DPAD + f] + b[f];
    v = fmaxf(v, 0.f);
    xout[(size_t)row * D + f] = v * inv;
  }
}

// ---------------------------------------------------------------------------
// Final MLP (single block)
// ---------------------------------------------------------------------------
__global__ void mlp_kernel(const float* __restrict__ x1, const float* __restrict__ x2,
                           const int* __restrict__ p1p, const int* __restrict__ p2p,
                           const float* __restrict__ W1, const float* __restrict__ b1,
                           const float* __restrict__ W2, const float* __restrict__ b2,
                           const float* __restrict__ W3, const float* __restrict__ b3,
                           float* __restrict__ score)
{
  __shared__ float h[600], h2[600], h3[300];
  int tid = threadIdx.x;
  int p1 = p1p[0], p2 = p2p[0];
  for (int f = tid; f < 300; f += 256) {
    h[f]       = x1[(size_t)p1 * D + f];
    h[300 + f] = x2[(size_t)p2 * D + f];
  }
  __syncthreads();
  for (int o = tid; o < 600; o += 256) {
    float acc = b1[o];
    for (int k = 0; k < 600; ++k) acc += h[k] * W1[(size_t)k * 600 + o];
    h2[o] = fmaxf(acc, 0.f);
  }
  __syncthreads();
  for (int o = tid; o < 300; o += 256) {
    float acc = b2[o];
    for (int k = 0; k < 600; ++k) acc += h2[k] * W2[(size_t)k * 300 + o];
    h3[o] = fmaxf(acc, 0.f);
  }
  __syncthreads();
  if (tid == 0) {
    float acc = b3[0];
    for (int k = 0; k < 300; ++k) acc += h3[k] * W3[k];
    score[0] = 1.f / (1.f + expf(-acc));
  }
}

// ---------------------------------------------------------------------------
// Host orchestration
// ---------------------------------------------------------------------------
static void launch_gemm(hipStream_t st, const _Float16* A, const _Float16* B,
                        float* C, int M, int Kp)
{
  const int Ntiles = DPAD / 16;  // 19
  dim3 block(32, 4);
  dim3 grid(M / 16, (Ntiles + 3) / 4);
  hipLaunchKernelGGL(wmma_gemm, grid, block, 0, st, A, B, C, D, Ntiles, Kp, DPAD);
}

static void run_simple_gnn(hipStream_t st, const float* xin, float* xout,
                           int N, int E, const int* src, const int* dst,
                           const _Float16* Wh, const float* as, const float* ad,
                           const float* b, _Float16* xh, float* z, float* es, float* ed,
                           float* emax, float* denom, float* ebuf, float* exbuf, float* agg)
{
  hipLaunchKernelGGL(convert_x, dim3((N * KPAD + 255) / 256), dim3(256), 0, st, xin, xh, N, 0);
  launch_gemm(st, xh, Wh, z, N, KPAD);
  hipLaunchKernelGGL(rowdot, dim3(N), dim3(32), 0, st, z, as, ad, es, ed);
  hipLaunchKernelGGL(sparse_init, dim3(N), dim3(320), 0, st, agg, emax, denom, b);
  int Etot = E + N;
  hipLaunchKernelGGL(edge_max, dim3((Etot + 255) / 256), dim3(256), 0, st, src, dst, E, N, es, ed, ebuf, emax);
  hipLaunchKernelGGL(edge_exp, dim3((Etot + 255) / 256), dim3(256), 0, st, src, dst, E, N, ebuf, emax, exbuf, denom);
  hipLaunchKernelGGL(edge_agg, dim3(Etot), dim3(128), 0, st, src, dst, E, exbuf, denom, z, agg);
  hipLaunchKernelGGL(finish_sparse, dim3(N), dim3(32), 0, st, agg, xout);
}

static void run_cross_gnn(hipStream_t st, const float* xa, int Na, const float* xb, int Nb,
                          float* xout, const _Float16* Wh, const float* as, const float* ad,
                          const float* b, _Float16* xh, float* z, float* es, float* ed,
                          _Float16* alphaH, float* aself, _Float16* zT, float* gout)
{
  // xc = concat(xa, xb); dst = rows [0,Na), src = rows [Na, Na+Nb)
  hipLaunchKernelGGL(convert_x, dim3((Na * KPAD + 255) / 256), dim3(256), 0, st, xa, xh, Na, 0);
  hipLaunchKernelGGL(convert_x, dim3((Nb * KPAD + 255) / 256), dim3(256), 0, st, xb, xh, Nb, Na);
  int Ntot = Na + Nb;
  launch_gemm(st, xh, Wh, z, Ntot, KPAD);
  hipLaunchKernelGGL(rowdot, dim3(Ntot), dim3(32), 0, st, z, as, ad, es, ed);
  hipLaunchKernelGGL(cross_softmax, dim3(Na), dim3(256), 0, st, es, ed, alphaH, aself, Nb, Na);
  hipLaunchKernelGGL(transpose_z, dim3((DPAD * Nb + 255) / 256), dim3(256), 0, st, z, zT, Nb, Na);
  launch_gemm(st, alphaH, zT, gout, Na, Nb);   // gout = alpha @ z_src
  hipLaunchKernelGGL(cross_finish, dim3(Na), dim3(32), 0, st, gout, z, aself, b, xout);
}

extern "C" void kernel_launch(void* const* d_in, const int* in_sizes, int n_in,
                              void* d_out, int out_size, void* d_ws, size_t ws_size,
                              hipStream_t stream)
{
  const int N1 = 1024, N2 = 512, E1 = 32768, E2 = 16384;

  const float* x1_in = (const float*)d_in[0];
  const float* x2_in = (const float*)d_in[1];
  const int* ei1 = (const int*)d_in[2];
  const int* ei2 = (const int*)d_in[3];
  const int* p1  = (const int*)d_in[6];
  const int* p2  = (const int*)d_in[7];
  const float* W_ts = (const float*)d_in[10]; const float* as_ts = (const float*)d_in[11];
  const float* ad_ts = (const float*)d_in[12]; const float* b_ts = (const float*)d_in[13];
  const float* W_gs = (const float*)d_in[14]; const float* as_gs = (const float*)d_in[15];
  const float* ad_gs = (const float*)d_in[16]; const float* b_gs = (const float*)d_in[17];
  const float* W_tc = (const float*)d_in[18]; const float* as_tc = (const float*)d_in[19];
  const float* ad_tc = (const float*)d_in[20]; const float* b_tc = (const float*)d_in[21];
  const float* W_gc = (const float*)d_in[22]; const float* as_gc = (const float*)d_in[23];
  const float* ad_gc = (const float*)d_in[24]; const float* b_gc = (const float*)d_in[25];
  const float* mW1 = (const float*)d_in[26]; const float* mb1 = (const float*)d_in[27];
  const float* mW2 = (const float*)d_in[28]; const float* mb2 = (const float*)d_in[29];
  const float* mW3 = (const float*)d_in[30]; const float* mb3 = (const float*)d_in[31];

  // ---- workspace carve ----
  char* base = (char*)d_ws;
  size_t off = 0;
  auto alloc = [&](size_t bytes) -> void* {
    off = (off + 255) & ~(size_t)255;
    void* p = base + off;
    off += bytes;
    return p;
  };
  float*    x1_cur = (float*)alloc((size_t)N1 * D * 4);
  float*    x1_mid = (float*)alloc((size_t)N1 * D * 4);
  float*    x2_cur = (float*)alloc((size_t)N2 * D * 4);
  float*    x2_mid = (float*)alloc((size_t)N2 * D * 4);
  _Float16* Wh_ts  = (_Float16*)alloc((size_t)DPAD * KPAD * 2);
  _Float16* Wh_gs  = (_Float16*)alloc((size_t)DPAD * KPAD * 2);
  _Float16* Wh_tc  = (_Float16*)alloc((size_t)DPAD * KPAD * 2);
  _Float16* Wh_gc  = (_Float16*)alloc((size_t)DPAD * KPAD * 2);
  _Float16* xh     = (_Float16*)alloc((size_t)(N1 + N2) * KPAD * 2);
  float*    z      = (float*)alloc((size_t)(N1 + N2) * DPAD * 4);
  float*    es     = (float*)alloc((size_t)(N1 + N2) * 4);
  float*    ed     = (float*)alloc((size_t)(N1 + N2) * 4);
  float*    emax   = (float*)alloc((size_t)N1 * 4);
  float*    denom  = (float*)alloc((size_t)N1 * 4);
  float*    aself  = (float*)alloc((size_t)N1 * 4);
  float*    ebuf   = (float*)alloc((size_t)(E1 + N1) * 4);
  float*    exbuf  = (float*)alloc((size_t)(E1 + N1) * 4);
  _Float16* alphaH = (_Float16*)alloc((size_t)N1 * N2 * 2 * 2);   // covers both cross dirs
  _Float16* zT     = (_Float16*)alloc((size_t)DPAD * N1 * 2);
  float*    gout   = (float*)alloc((size_t)N1 * DPAD * 4);
  float*    agg    = (float*)alloc((size_t)N1 * DPAD * 4);

  // ---- one-time weight conversion (per launch; deterministic) ----
  const int wcnt = (DPAD * KPAD + 255) / 256;
  hipLaunchKernelGGL(convert_W, dim3(wcnt), dim3(256), 0, stream, W_ts, Wh_ts);
  hipLaunchKernelGGL(convert_W, dim3(wcnt), dim3(256), 0, stream, W_gs, Wh_gs);
  hipLaunchKernelGGL(convert_W, dim3(wcnt), dim3(256), 0, stream, W_tc, Wh_tc);
  hipLaunchKernelGGL(convert_W, dim3(wcnt), dim3(256), 0, stream, W_gc, Wh_gc);

  hipMemcpyAsync(x1_cur, x1_in, (size_t)N1 * D * 4, hipMemcpyDeviceToDevice, stream);
  hipMemcpyAsync(x2_cur, x2_in, (size_t)N2 * D * 4, hipMemcpyDeviceToDevice, stream);

  for (int it = 0; it < 16; ++it) {
    hipLaunchKernelGGL(l2norm_inplace, dim3(N1), dim3(32), 0, stream, x1_cur);
    hipLaunchKernelGGL(l2norm_inplace, dim3(N2), dim3(32), 0, stream, x2_cur);

    run_simple_gnn(stream, x1_cur, x1_mid, N1, E1, ei1, ei1 + E1,
                   Wh_ts, as_ts, ad_ts, b_ts, xh, z, es, ed,
                   emax, denom, ebuf, exbuf, agg);
    run_simple_gnn(stream, x2_cur, x2_mid, N2, E2, ei2, ei2 + E2,
                   Wh_gs, as_gs, ad_gs, b_gs, xh, z, es, ed,
                   emax, denom, ebuf, exbuf, agg);

    // cross 1: concat(x1, x2), dst = x1 nodes
    run_cross_gnn(stream, x1_mid, N1, x2_mid, N2, x1_cur,
                  Wh_tc, as_tc, ad_tc, b_tc, xh, z, es, ed, alphaH, aself, zT, gout);
    // cross 2: concat(x2, x1), dst = x2 nodes
    run_cross_gnn(stream, x2_mid, N2, x1_mid, N1, x2_cur,
                  Wh_gc, as_gc, ad_gc, b_gc, xh, z, es, ed, alphaH, aself, zT, gout);
  }

  float* out = (float*)d_out;
  hipMemcpyAsync(out, x1_cur, (size_t)N1 * D * 4, hipMemcpyDeviceToDevice, stream);
  hipMemcpyAsync(out + (size_t)N1 * D, x2_cur, (size_t)N2 * D * 4,
                 hipMemcpyDeviceToDevice, stream);
  hipLaunchKernelGGL(mlp_kernel, dim3(1), dim3(256), 0, stream,
                     x1_cur, x2_cur, p1, p2, mW1, mb1, mW2, mb2, mW3, mb3,
                     out + (size_t)N1 * D + (size_t)N2 * D);
}
